// DDRec_40544491274539
// MI455X (gfx1250) — matrix operand order repaired and checked by reading, not compiled, not run on hardware
//
#include <hip/hip_runtime.h>
#include <hip/hip_bf16.h>

// ---------------- problem constants (match reference) ----------------
#define NU_   8000
#define NI_   5000
#define NT_   13000      // NU + NI
#define E_    400000
#define DIM_  64
#define DV_   512
#define DT_   256
#define L_    2
#define K_    10
#define OUTC_ 192        // 3*DIM
#define MMVAL_ (1.0f/(10.0f+1e-7f))   // deg == K exactly -> constant spmm weight

typedef __bf16 bf16_t;
typedef __attribute__((ext_vector_type(16))) __bf16 v16bf;
typedef __attribute__((ext_vector_type(8)))  __bf16 bf16x8;
typedef __attribute__((ext_vector_type(8)))  float  v8f;

static inline int cdiv(int a, int b) { return (a + b - 1) / b; }

// ---------------- WMMA fragment loaders (wave32, ISA 7.12.2 layouts) --
// 16-bit A 16x32: lanes 0-15 row M=lane, K = kb+0..7 (v0..3) and kb+16..23 (v4..7)
//                 lanes 16-31 row M=lane-16, K = kb+8..15 and kb+24..31
// Row is CLAMPED (no divergence in inner loop); OOB rows are store-guarded.
__device__ __forceinline__ v16bf load_a_frag(const bf16_t* A, int lda, int rowsMax,
                                             int row0, int kb, int lane) {
  int r = row0 + (lane & 15);
  if (r >= rowsMax) r = rowsMax - 1;
  int half = lane >> 4;
  const bf16_t* p = A + (size_t)r * lda + kb + half * 8;
  bf16x8 lo = *(const bf16x8*)(p);
  bf16x8 hi = *(const bf16x8*)(p + 16);
  v16bf a;
#pragma unroll
  for (int i = 0; i < 8; ++i) { a[i] = lo[i]; a[8 + i] = hi[i]; }
  return a;
}

// Same A layout, but reading a 16-row strip resident in LDS (row-major, leading dim = ldk).
__device__ __forceinline__ v16bf load_a_frag_lds(const bf16_t* shA, int ldk, int kb, int lane) {
  int r = lane & 15;
  int half = lane >> 4;
  const bf16_t* p = shA + r * ldk + kb + half * 8;
  bf16x8 lo = *(const bf16x8*)(p);
  bf16x8 hi = *(const bf16x8*)(p + 16);
  v16bf a;
#pragma unroll
  for (int i = 0; i < 8; ++i) { a[i] = lo[i]; a[8 + i] = hi[i]; }
  return a;
}

// 16-bit B 32x16: lane holds column N=lane&15; lanes 0-15 K=kb+0..7,kb+16..23, lanes 16-31 K=kb+8..15,kb+24..31
__device__ __forceinline__ v16bf load_b_frag(const bf16_t* B, int ldb, int colsMax,
                                             int col0, int kb, int lane) {
  int c = col0 + (lane & 15);
  if (c >= colsMax) c = colsMax - 1;  // clamp; consumers guard these columns
  int half = lane >> 4;
  const bf16_t* p = B + (size_t)c * ldb + kb + half * 8;
  bf16x8 lo = *(const bf16x8*)(p);
  bf16x8 hi = *(const bf16x8*)(p + 16);
  v16bf b;
#pragma unroll
  for (int i = 0; i < 8; ++i) { b[i] = lo[i]; b[8 + i] = hi[i]; }
  return b;
}

#define WMMA_BF16(a, b, c) \
  __builtin_amdgcn_wmma_f32_16x16x32_bf16(false, (a), false, (b), (short)0, (c), false, false)

// ---------------- elementwise helpers ----------------
__global__ void k_cast_bf16(const float* __restrict__ in, bf16_t* __restrict__ out, int n) {
  int i = blockIdx.x * blockDim.x + threadIdx.x;
  if (i < n) out[i] = (bf16_t)in[i];
}

__global__ void k_fill_f32(float* __restrict__ p, float v, int n) {
  int i = blockIdx.x * blockDim.x + threadIdx.x;
  if (i < n) p[i] = v;
}

__global__ void k_copy_f32(const float* __restrict__ a, float* __restrict__ b, int n) {
  int i = blockIdx.x * blockDim.x + threadIdx.x;
  if (i < n) b[i] = a[i];
}

// L2-normalize each row, emit bf16
__global__ void k_rownorm_bf16(const float* __restrict__ feat, bf16_t* __restrict__ out,
                               int rows, int d) {
  int row  = blockIdx.x * (blockDim.x >> 5) + (threadIdx.x >> 5);
  int lane = threadIdx.x & 31;
  if (row >= rows) return;
  const float* f = feat + (size_t)row * d;
  float s = 0.f;
  for (int k = lane; k < d; k += 32) { float v = f[k]; s += v * v; }
#pragma unroll
  for (int off = 16; off > 0; off >>= 1) s += __shfl_xor(s, off, 32);
  float rinv = rsqrtf(s + 1e-20f);
  bf16_t* o = out + (size_t)row * d;
  for (int k = lane; k < d; k += 32) o[k] = (bf16_t)(f[k] * rinv);
}

// ------- WMMA GEMM, Nout fixed to 64: C[m][0..63] = sum_k A[m][k]*B[n][k] (+bias[n])
// One wave computes a 16x64 output strip: A fragment reused across 4 N-tiles.
__global__ void k_wmma_gemm_n64(const bf16_t* __restrict__ A, const bf16_t* __restrict__ B,
                                const float* __restrict__ bias, float* __restrict__ C,
                                int M, int Kd) {
  int wave = (int)((blockIdx.x * blockDim.x + threadIdx.x) >> 5);
  int lane = threadIdx.x & 31;
  int row0 = wave * 16;
  if (row0 >= M) return;
  v8f c0 = {}, c1 = {}, c2 = {}, c3 = {};
  for (int kb = 0; kb < Kd; kb += 32) {
    v16bf a  = load_a_frag(A, Kd, M, row0, kb, lane);
    v16bf b0 = load_b_frag(B, Kd, 64,  0, kb, lane);
    v16bf b1 = load_b_frag(B, Kd, 64, 16, kb, lane);
    v16bf b2 = load_b_frag(B, Kd, 64, 32, kb, lane);
    v16bf b3 = load_b_frag(B, Kd, 64, 48, kb, lane);
    c0 = WMMA_BF16(a, b0, c0);
    c1 = WMMA_BF16(a, b1, c1);
    c2 = WMMA_BF16(a, b2, c2);
    c3 = WMMA_BF16(a, b3, c3);
  }
  int n = lane & 15, half = lane >> 4;
  float bv0 = bias ? bias[n]      : 0.0f;
  float bv1 = bias ? bias[n + 16] : 0.0f;
  float bv2 = bias ? bias[n + 32] : 0.0f;
  float bv3 = bias ? bias[n + 48] : 0.0f;
#pragma unroll
  for (int i = 0; i < 8; ++i) {
    int m = row0 + half * 8 + i;
    if (m < M) {
      float* r = C + (size_t)m * 64;
      r[n]      = c0[i] + bv0;
      r[n + 16] = c1[i] + bv1;
      r[n + 32] = c2[i] + bv2;
      r[n + 48] = c3[i] + bv3;
    }
  }
}

// ---------------- top-K helper ----------------
__device__ __forceinline__ void topk_insert(float v, int idx, float* topv, int* topi) {
  int mi = 0; float mv = topv[0];
#pragma unroll
  for (int k = 1; k < K_; ++k) if (topv[k] < mv) { mv = topv[k]; mi = k; }
  if (v > mv) { topv[mi] = v; topi[mi] = idx; }
}

// -------- fused similarity + top-K: one wave per 16-row strip --------
// A strip (16 x KD bf16) staged into LDS ONCE via async global->LDS DMA,
// then reused across all 5000 columns (79 blocks of 64 cols, 4 WMMA accumulators).
template <int KD>
__global__ void k_sim_topk(const bf16_t* __restrict__ FN, int* __restrict__ knn_idx) {
  __shared__ bf16_t shA[16 * KD];
  __shared__ float  tile[16][72];     // 16x64 scores + pad
  int lane = threadIdx.x;             // blockDim == 32
  int row0 = blockIdx.x * 16;

  // ---- async stage of A strip: 16 rows * KD bf16 = (2*KD) 16B chunks ----
  {
    const int NCHUNK = 2 * KD;        // 16*KD*2 bytes / 16
    unsigned lds_base = (unsigned)(unsigned long long)(size_t)(&shA[0]);
    for (int ch = lane; ch < NCHUNK; ch += 32) {
      int elt = ch * 8;               // bf16 element index within strip
      int r   = elt / KD;
      int kk  = elt - r * KD;
      int rr  = row0 + r; if (rr >= NI_) rr = NI_ - 1;
      unsigned long long gaddr = (unsigned long long)(size_t)(FN + (size_t)rr * KD + kk);
      unsigned lds_off = lds_base + (unsigned)(elt * 2);
      asm volatile("global_load_async_to_lds_b128 %0, %1, off"
                   :: "v"(lds_off), "v"(gaddr) : "memory");
    }
    asm volatile("s_wait_asynccnt 0x0" ::: "memory");
    __syncthreads();
  }

  float topv[K_]; int topi[K_];
#pragma unroll
  for (int k = 0; k < K_; ++k) { topv[k] = -1e30f; topi[k] = 0; }
  int r16   = lane & 15;
  int myrow = row0 + r16;

  for (int j0 = 0; j0 < NI_; j0 += 64) {
    // prefetch next column block's rows (speculative, L2)
    int pj = j0 + 64 + r16;
    if (pj < NI_) __builtin_prefetch(FN + (size_t)pj * KD, 0, 1);

    v8f c0 = {}, c1 = {}, c2 = {}, c3 = {};
    for (int kb = 0; kb < KD; kb += 32) {
      v16bf a  = load_a_frag_lds(shA, KD, kb, lane);
      v16bf b0 = load_b_frag(FN, KD, NI_, j0,      kb, lane);
      v16bf b1 = load_b_frag(FN, KD, NI_, j0 + 16, kb, lane);
      v16bf b2 = load_b_frag(FN, KD, NI_, j0 + 32, kb, lane);
      v16bf b3 = load_b_frag(FN, KD, NI_, j0 + 48, kb, lane);
      c0 = WMMA_BF16(a, b0, c0);
      c1 = WMMA_BF16(a, b1, c1);
      c2 = WMMA_BF16(a, b2, c2);
      c3 = WMMA_BF16(a, b3, c3);
    }
    int n = lane & 15, half = lane >> 4;
#pragma unroll
    for (int i = 0; i < 8; ++i) {
      int mr = half * 8 + i;
      tile[mr][n]      = c0[i];
      tile[mr][n + 16] = c1[i];
      tile[mr][n + 32] = c2[i];
      tile[mr][n + 48] = c3[i];
    }
    __syncthreads();
    // lane r scans cols 0..31 of row r; lane r+16 scans cols 32..63 of row r
    if (myrow < NI_) {
      int cbase = (lane >> 4) * 32;
      for (int j = 0; j < 32; ++j) {
        int col = j0 + cbase + j;
        if (col < NI_) topk_insert(tile[r16][cbase + j], col, topv, topi);
      }
    }
    __syncthreads();
  }

  // merge upper half-lane's top-K into lower half-lane's, then write
#pragma unroll
  for (int k = 0; k < K_; ++k) {
    float ov = __shfl(topv[k], lane + 16, 32);
    int   oi = __shfl(topi[k], lane + 16, 32);
    if (lane < 16) topk_insert(ov, oi, topv, topi);
  }
  if (lane < 16 && myrow < NI_) {
#pragma unroll
    for (int k = 0; k < K_; ++k) knn_idx[myrow * K_ + k] = topi[k];
  }
}

// ---------------- graph kernels ----------------
__global__ void k_build_ego(const float* __restrict__ user_emb, const float* __restrict__ item_part,
                            float* __restrict__ ego) {
  int idx = blockIdx.x * blockDim.x + threadIdx.x;
  if (idx >= NT_ * DIM_) return;
  int n = idx / DIM_, d = idx % DIM_;
  ego[idx] = (n < NU_) ? user_emb[idx] : item_part[(size_t)(n - NU_) * DIM_ + d];
}

__global__ void k_edge_mask(const float* __restrict__ ego, const int* __restrict__ eu,
                            const int* __restrict__ ei, float* __restrict__ m) {
  int e = blockIdx.x * blockDim.x + threadIdx.x;
  if (e >= E_) return;
  const float4* a = (const float4*)(ego + (size_t)eu[e] * DIM_);
  const float4* b = (const float4*)(ego + (size_t)(ei[e] + NU_) * DIM_);
  float s = 0.f;
#pragma unroll
  for (int k = 0; k < DIM_ / 4; ++k) {
    float4 x = a[k], y = b[k];
    s += x.x * y.x + x.y * y.y + x.z * y.z + x.w * y.w;
  }
  m[e] = (s >= 0.0f) ? 1.0f : 0.0f;
}

__global__ void k_deg_accum(const float* __restrict__ m, const int* __restrict__ eu,
                            const int* __restrict__ ei, float* __restrict__ deg) {
  int e = blockIdx.x * blockDim.x + threadIdx.x;
  if (e >= E_) return;
  float w = m ? m[e] : 1.0f;
  if (w != 0.0f) {
    atomicAdd(&deg[eu[e]], w);
    atomicAdd(&deg[ei[e] + NU_], w);
  }
}

__global__ void k_dinv(float* __restrict__ deg) {
  int n = blockIdx.x * blockDim.x + threadIdx.x;
  if (n < NT_) deg[n] = rsqrtf(deg[n] + 1.0f);
}

// coef = dinv[src]*dinv[dst]*w is symmetric; each thread handles 1 edge, 2 dims, both directions
__global__ void k_agg_scatter(const float* __restrict__ m, const int* __restrict__ eu,
                              const int* __restrict__ ei, const float* __restrict__ dinv,
                              const float* __restrict__ xw, float* __restrict__ agg) {
  long long t = (long long)blockIdx.x * blockDim.x + threadIdx.x;
  int e = (int)(t >> 5);
  if (e >= E_) return;
  int l = (int)(t & 31);
  float w = m ? m[e] : 1.0f;
  if (w == 0.0f) return;
  int u = eu[e], it = ei[e] + NU_;
  float coef = dinv[u] * dinv[it] * w;
  const float* xu = xw + (size_t)u * DIM_;
  const float* xi = xw + (size_t)it * DIM_;
  float* au = agg + (size_t)u * DIM_;
  float* ai = agg + (size_t)it * DIM_;
  atomicAdd(&ai[l],       coef * xu[l]);
  atomicAdd(&ai[l + 32],  coef * xu[l + 32]);
  atomicAdd(&au[l],       coef * xi[l]);
  atomicAdd(&au[l + 32],  coef * xi[l + 32]);
}

__global__ void k_gcn_finalize(float* __restrict__ ego, float* __restrict__ acc,
                               const float* __restrict__ agg, const float* __restrict__ xw,
                               const float* __restrict__ dinv, const float* __restrict__ b) {
  int idx = blockIdx.x * blockDim.x + threadIdx.x;
  if (idx >= NT_ * DIM_) return;
  int n = idx / DIM_, d = idx % DIM_;
  float di = dinv[n];
  float v = agg[idx] + di * di * xw[idx] + b[d];
  ego[idx] = v;
  acc[idx] += v;
}

// ---------------- output assembly ----------------
__global__ void k_write_user(const float* __restrict__ acc, float* __restrict__ out, int col) {
  int idx = blockIdx.x * blockDim.x + threadIdx.x;
  if (idx >= NU_ * DIM_) return;
  int n = idx / DIM_, d = idx % DIM_;
  out[(size_t)n * OUTC_ + col + d] = acc[idx] * (1.0f / 3.0f);
}

__global__ void k_write_item_scratch(const float* __restrict__ acc, float* __restrict__ iX) {
  int idx = blockIdx.x * blockDim.x + threadIdx.x;
  if (idx >= NI_ * DIM_) return;
  iX[idx] = acc[(size_t)NU_ * DIM_ + idx] * (1.0f / 3.0f);
}

__global__ void k_mm_write(const float* __restrict__ base, const int* __restrict__ kv,
                           const int* __restrict__ kt, float* __restrict__ out, int col) {
  int idx = blockIdx.x * blockDim.x + threadIdx.x;
  if (idx >= NI_ * DIM_) return;
  int i = idx / DIM_, d = idx % DIM_;
  float sv = 0.f, st = 0.f;
#pragma unroll
  for (int k = 0; k < K_; ++k) {
    sv += base[(size_t)kv[i * K_ + k] * DIM_ + d];
    st += base[(size_t)kt[i * K_ + k] * DIM_ + d];
  }
  float r = base[idx] + 0.5f * MMVAL_ * sv + 0.5f * MMVAL_ * st;
  out[(size_t)(NU_ + i) * OUTC_ + col + d] = r;
}

// ---------------- host orchestration ----------------
static void gemm_n64(const bf16_t* A, const bf16_t* B, const float* bias, float* C,
                     int M, int Kd, hipStream_t s) {
  int waves  = cdiv(M, 16);
  int blocks = cdiv(waves * 32, 256);
  k_wmma_gemm_n64<<<blocks, 256, 0, s>>>(A, B, bias, C, M, Kd);
}

static void run_prop(const float* user_emb, const float* item_part, bool filtered,
                     const bf16_t* wgcn_bf, const float* gcn_b,
                     const int* eu, const int* ei,
                     float* ego, bf16_t* ego_bf, float* xw, float* agg, float* acc,
                     float* deg, float* mbuf, hipStream_t s) {
  const int ND = NT_ * DIM_;
  k_build_ego<<<cdiv(ND, 256), 256, 0, s>>>(user_emb, item_part, ego);
  k_copy_f32<<<cdiv(ND, 256), 256, 0, s>>>(ego, acc, ND);
  for (int l = 0; l < L_; ++l) {
    const float* mptr = nullptr;
    if (filtered) {
      k_edge_mask<<<cdiv(E_, 256), 256, 0, s>>>(ego, eu, ei, mbuf);
      mptr = mbuf;
    }
    k_fill_f32<<<cdiv(NT_, 256), 256, 0, s>>>(deg, 0.0f, NT_);
    k_deg_accum<<<cdiv(E_, 256), 256, 0, s>>>(mptr, eu, ei, deg);
    k_dinv<<<cdiv(NT_, 256), 256, 0, s>>>(deg);   // deg now holds dinv
    k_cast_bf16<<<cdiv(ND, 256), 256, 0, s>>>(ego, ego_bf, ND);
    gemm_n64(ego_bf, wgcn_bf + (size_t)l * DIM_ * DIM_, nullptr, xw, NT_, DIM_, s);
    k_fill_f32<<<cdiv(ND, 256), 256, 0, s>>>(agg, 0.0f, ND);
    {
      long long threads = (long long)E_ * 32;
      k_agg_scatter<<<(int)((threads + 255) / 256), 256, 0, s>>>(mptr, eu, ei, deg, xw, agg);
    }
    k_gcn_finalize<<<cdiv(ND, 256), 256, 0, s>>>(ego, acc, agg, xw, deg, gcn_b + (size_t)l * DIM_);
  }
}

extern "C" void kernel_launch(void* const* d_in, const int* in_sizes, int n_in,
                              void* d_out, int out_size, void* d_ws, size_t ws_size,
                              hipStream_t stream) {
  (void)in_sizes; (void)n_in; (void)out_size; (void)ws_size;
  const float* user_emb = (const float*)d_in[0];
  const float* item_emb = (const float*)d_in[1];
  const float* v_feat   = (const float*)d_in[2];
  const float* t_feat   = (const float*)d_in[3];
  const float* img_w    = (const float*)d_in[4];
  const float* img_b    = (const float*)d_in[5];
  const float* txt_w    = (const float*)d_in[6];
  const float* txt_b    = (const float*)d_in[7];
  const float* gcn_w    = (const float*)d_in[8];
  const float* gcn_b    = (const float*)d_in[9];
  const int*   eu       = (const int*)d_in[10];
  const int*   ei       = (const int*)d_in[11];
  float* out = (float*)d_out;

  // bump allocator over workspace (256B aligned slots)
  char* base = (char*)d_ws;
  size_t off = 0;
  auto alloc = [&](size_t bytes) -> void* {
    void* p = base + off;
    off = (off + bytes + 255) & ~(size_t)255;
    return p;
  };
  bf16_t* vfeat_bf = (bf16_t*)alloc((size_t)NI_ * DV_ * 2);
  bf16_t* tfeat_bf = (bf16_t*)alloc((size_t)NI_ * DT_ * 2);
  bf16_t* fnv_bf   = (bf16_t*)alloc((size_t)NI_ * DV_ * 2);
  bf16_t* fnt_bf   = (bf16_t*)alloc((size_t)NI_ * DT_ * 2);
  bf16_t* wimg_bf  = (bf16_t*)alloc((size_t)DIM_ * DV_ * 2);
  bf16_t* wtxt_bf  = (bf16_t*)alloc((size_t)DIM_ * DT_ * 2);
  bf16_t* wgcn_bf  = (bf16_t*)alloc((size_t)L_ * DIM_ * DIM_ * 2);
  float*  v_emb    = (float*)alloc((size_t)NI_ * DIM_ * 4);
  float*  t_emb    = (float*)alloc((size_t)NI_ * DIM_ * 4);
  float*  ego      = (float*)alloc((size_t)NT_ * DIM_ * 4);
  bf16_t* ego_bf   = (bf16_t*)alloc((size_t)NT_ * DIM_ * 2);
  float*  xw       = (float*)alloc((size_t)NT_ * DIM_ * 4);
  float*  agg      = (float*)alloc((size_t)NT_ * DIM_ * 4);
  float*  acc      = (float*)alloc((size_t)NT_ * DIM_ * 4);
  float*  deg      = (float*)alloc((size_t)NT_ * 4);
  float*  mbuf     = (float*)alloc((size_t)E_ * 4);
  int*    knn_v    = (int*)alloc((size_t)NI_ * K_ * 4);
  int*    knn_t    = (int*)alloc((size_t)NI_ * K_ * 4);
  float*  iG       = (float*)alloc((size_t)NI_ * DIM_ * 4);
  float*  iV       = (float*)alloc((size_t)NI_ * DIM_ * 4);
  float*  iT       = (float*)alloc((size_t)NI_ * DIM_ * 4);

  // ---- bf16 packing ----
  k_cast_bf16<<<cdiv(NI_ * DV_, 256), 256, 0, stream>>>(v_feat, vfeat_bf, NI_ * DV_);
  k_cast_bf16<<<cdiv(NI_ * DT_, 256), 256, 0, stream>>>(t_feat, tfeat_bf, NI_ * DT_);
  k_cast_bf16<<<cdiv(DIM_ * DV_, 256), 256, 0, stream>>>(img_w, wimg_bf, DIM_ * DV_);
  k_cast_bf16<<<cdiv(DIM_ * DT_, 256), 256, 0, stream>>>(txt_w, wtxt_bf, DIM_ * DT_);
  k_cast_bf16<<<cdiv(L_ * DIM_ * DIM_, 256), 256, 0, stream>>>(gcn_w, wgcn_bf, L_ * DIM_ * DIM_);

  // ---- KNN graphs: normalize rows, async-staged WMMA similarity + fused top-K ----
  k_rownorm_bf16<<<cdiv(NI_, 8), 256, 0, stream>>>(v_feat, fnv_bf, NI_, DV_);
  k_rownorm_bf16<<<cdiv(NI_, 8), 256, 0, stream>>>(t_feat, fnt_bf, NI_, DT_);
  k_sim_topk<DV_><<<cdiv(NI_, 16), 32, 0, stream>>>(fnv_bf, knn_v);
  k_sim_topk<DT_><<<cdiv(NI_, 16), 32, 0, stream>>>(fnt_bf, knn_t);

  // ---- modality transforms (WMMA GEMM + bias) ----
  gemm_n64(vfeat_bf, wimg_bf, img_b, v_emb, NI_, DV_, stream);
  gemm_n64(tfeat_bf, wtxt_bf, txt_b, t_emb, NI_, DT_, stream);

  // ---- prop(visual, filtered) ----
  run_prop(user_emb, v_emb, true, wgcn_bf, gcn_b, eu, ei,
           ego, ego_bf, xw, agg, acc, deg, mbuf, stream);
  k_write_user<<<cdiv(NU_ * DIM_, 256), 256, 0, stream>>>(acc, out, 64);
  k_write_item_scratch<<<cdiv(NI_ * DIM_, 256), 256, 0, stream>>>(acc, iV);

  // ---- prop(text, filtered) ----
  run_prop(user_emb, t_emb, true, wgcn_bf, gcn_b, eu, ei,
           ego, ego_bf, xw, agg, acc, deg, mbuf, stream);
  k_write_user<<<cdiv(NU_ * DIM_, 256), 256, 0, stream>>>(acc, out, 128);
  k_write_item_scratch<<<cdiv(NI_ * DIM_, 256), 256, 0, stream>>>(acc, iT);

  // ---- prop(id embeddings, unfiltered) ----
  run_prop(user_emb, item_emb, false, wgcn_bf, gcn_b, eu, ei,
           ego, ego_bf, xw, agg, acc, deg, mbuf, stream);
  k_write_user<<<cdiv(NU_ * DIM_, 256), 256, 0, stream>>>(acc, out, 0);
  k_write_item_scratch<<<cdiv(NI_ * DIM_, 256), 256, 0, stream>>>(acc, iG);

  // ---- mm_prop (constant-val SPMM over KNN graphs) + item output ----
  k_mm_write<<<cdiv(NI_ * DIM_, 256), 256, 0, stream>>>(iG, knn_v, knn_t, out, 0);
  k_mm_write<<<cdiv(NI_ * DIM_, 256), 256, 0, stream>>>(iV, knn_v, knn_t, out, 64);
  k_mm_write<<<cdiv(NI_ * DIM_, 256), 256, 0, stream>>>(iT, knn_v, knn_t, out, 128);
}
